// GATTemporalModel_23742579212305
// MI455X (gfx1250) — matrix operand order repaired and checked by reading, not compiled, not run on hardware
//
#include <hip/hip_runtime.h>
#include <hip/hip_bf16.h>

// ---------------------------------------------------------------------------
// GAT(2 layers) + temporal scatter + 2-layer LSTM + FC for MI455X (gfx1250).
// Matrix ops use v_wmma_f32_16x16x32_f16, compile-time tile shapes.
// LSTM recurrence: persistent single-WGP workgroup; h state AND Whh^T staged
// in LDS (64KB of the 320KB WGP LDS) so nothing spills to scratch; G loads
// use one base address + immediate offsets (T=50 folded as a constant).
// ---------------------------------------------------------------------------

typedef __attribute__((ext_vector_type(16))) _Float16 v16h;
typedef __attribute__((ext_vector_type(8)))  _Float16 v8h;
typedef __attribute__((ext_vector_type(4)))  _Float16 v4h;
typedef __attribute__((ext_vector_type(8)))  float    v8f;

__device__ __forceinline__ v8f v8f_zero() {
  v8f z;
#pragma unroll
  for (int i = 0; i < 8; ++i) z[i] = 0.f;
  return z;
}

// fast branch-free transcendentals (v_exp_f32 / v_tanh_f32)
__device__ __forceinline__ float fexp2_(float x) {
#if __has_builtin(__builtin_amdgcn_exp2f)
  return __builtin_amdgcn_exp2f(x);
#else
  return exp2f(x);
#endif
}
__device__ __forceinline__ float fexp_(float x) {
  return fexp2_(x * 1.44269504088896340736f);
}
__device__ __forceinline__ float sigmoidf_(float x) {
  return 1.f / (1.f + fexp2_(x * -1.44269504088896340736f));
}
__device__ __forceinline__ float tanhf_(float x) {
#if __has_builtin(__builtin_amdgcn_tanhf)
  return __builtin_amdgcn_tanhf(x);
#else
  float e = fexp2_(x * 2.88539008177792681472f);   // exp(2x)
  return (e - 1.f) / (e + 1.f);
#endif
}

// monotone float <-> uint encoding so atomicMax(unsigned) == float max
__device__ __forceinline__ unsigned enc_f32(float f) {
  unsigned u = __float_as_uint(f);
  return (u & 0x80000000u) ? ~u : (u | 0x80000000u);
}
__device__ __forceinline__ float dec_f32(unsigned u) {
  unsigned b = (u & 0x80000000u) ? (u & 0x7fffffffu) : ~u;
  return __uint_as_float(b);
}

// ---------------------------------------------------------------------------
// zero fill (grid-stride)
// ---------------------------------------------------------------------------
__global__ void zero_u32(unsigned* __restrict__ p, long long n) {
  long long i = (long long)blockIdx.x * blockDim.x + threadIdx.x;
  long long stride = (long long)gridDim.x * blockDim.x;
  for (; i < n; i += stride) p[i] = 0u;
}

// ---------------------------------------------------------------------------
// GAT layer 1 node stage: xw1 = x@W1 [N,128]; es1/ed1 [N,2]. One wave / node.
// ---------------------------------------------------------------------------
__global__ void gat1_node(const float* __restrict__ x, const float* __restrict__ W1,
                          const float* __restrict__ a_src, const float* __restrict__ a_dst,
                          float* __restrict__ xw, float* __restrict__ es,
                          float* __restrict__ ed, int Nn) {
  int wv = blockIdx.x * 8 + (threadIdx.x >> 5);
  int lane = threadIdx.x & 31;
  if (wv >= Nn) return;
  float xv[6];
#pragma unroll
  for (int k = 0; k < 6; ++k) xv[k] = x[(long long)wv * 6 + k];
  int c0 = lane * 4;
  int head = lane >> 4;                 // 4*lane channels all in same head
  float ps = 0.f, pd = 0.f;
#pragma unroll
  for (int j = 0; j < 4; ++j) {
    float s = 0.f;
#pragma unroll
    for (int k = 0; k < 6; ++k) s += xv[k] * W1[k * 128 + c0 + j];
    xw[(long long)wv * 128 + c0 + j] = s;
    int cc = (c0 + j) & 63;
    ps += s * a_src[head * 64 + cc];
    pd += s * a_dst[head * 64 + cc];
  }
#pragma unroll
  for (int m = 1; m < 16; m <<= 1) {
    ps += __shfl_xor(ps, m, 32);
    pd += __shfl_xor(pd, m, 32);
  }
  if ((lane & 15) == 0) {
    es[wv * 2 + head] = ps;
    ed[wv * 2 + head] = pd;
  }
}

// ---------------------------------------------------------------------------
// GAT layer 2 attention scores from xw2 [N,64]. One wave / node.
// ---------------------------------------------------------------------------
__global__ void gat2_scores(const float* __restrict__ xw2, const float* __restrict__ a_src,
                            const float* __restrict__ a_dst, float* __restrict__ es,
                            float* __restrict__ ed, int Nn) {
  int wv = blockIdx.x * 8 + (threadIdx.x >> 5);
  int lane = threadIdx.x & 31;
  if (wv >= Nn) return;
  int c = lane * 2;
  float v0 = xw2[(long long)wv * 64 + c];
  float v1 = xw2[(long long)wv * 64 + c + 1];
  float ps = v0 * a_src[c] + v1 * a_src[c + 1];
  float pd = v0 * a_dst[c] + v1 * a_dst[c + 1];
#pragma unroll
  for (int m = 1; m < 32; m <<= 1) {
    ps += __shfl_xor(ps, m, 32);
    pd += __shfl_xor(pd, m, 32);
  }
  if (lane == 0) { es[wv] = ps; ed[wv] = pd; }
}

// ---------------------------------------------------------------------------
// Edge softmax passes (H compile-time: divisions become shifts/masks).
// ---------------------------------------------------------------------------
template <int H>
__global__ void edge_pass1(const int* __restrict__ ei, int E, int EP,
                           const float* __restrict__ es, const float* __restrict__ ed,
                           float* __restrict__ ebuf, unsigned* __restrict__ emax) {
  long long idx = (long long)blockIdx.x * blockDim.x + threadIdx.x;
  if (idx >= (long long)EP * H) return;
  int e = (int)(idx / H), h = (int)(idx % H);
  int s = e < E ? ei[e] : e - E;
  int d = e < E ? ei[E + e] : e - E;
  float v = es[(long long)s * H + h] + ed[(long long)d * H + h];
  v = v > 0.f ? v : 0.2f * v;
  ebuf[idx] = v;
  atomicMax(&emax[(long long)d * H + h], enc_f32(v));
}

template <int H>
__global__ void edge_pass2(const int* __restrict__ ei, int E, int EP,
                           float* __restrict__ ebuf, const unsigned* __restrict__ emax,
                           float* __restrict__ denom) {
  long long idx = (long long)blockIdx.x * blockDim.x + threadIdx.x;
  if (idx >= (long long)EP * H) return;
  int e = (int)(idx / H), h = (int)(idx % H);
  int d = e < E ? ei[E + e] : e - E;
  float ex = fexp_(ebuf[idx] - dec_f32(emax[(long long)d * H + h]));
  ebuf[idx] = ex;
  atomicAdd(&denom[(long long)d * H + h], ex);
}

// pass 3: acc[dst] += xw[src] * alpha   (TPE threads/edge, 4 channels each)
template <int H, int C, int TPE>
__global__ void edge_pass3(const int* __restrict__ ei, int E, int EP,
                           const float* __restrict__ ebuf, const float* __restrict__ denom,
                           const float* __restrict__ xw, float* __restrict__ acc) {
  long long idx = (long long)blockIdx.x * blockDim.x + threadIdx.x;
  if (idx >= (long long)EP * TPE) return;
  int e = (int)(idx / TPE);
  int c0 = (int)(idx % TPE) * 4;
  int s = e < E ? ei[e] : e - E;
  int d = e < E ? ei[E + e] : e - E;
  int h = c0 >> 6;
  float alpha = ebuf[(long long)e * H + h] / denom[(long long)d * H + h];
  const float4 v = *reinterpret_cast<const float4*>(xw + (long long)s * C + c0);
  float* ap = acc + (long long)d * C + c0;
  atomicAdd(ap + 0, v.x * alpha);
  atomicAdd(ap + 1, v.y * alpha);
  atomicAdd(ap + 2, v.z * alpha);
  atomicAdd(ap + 3, v.w * alpha);
}

// relu(acc + bias) -> f16 (feeds WMMA GEMM)
__global__ void finalize_relu_f16(const float* __restrict__ acc, const float* __restrict__ bias,
                                  _Float16* __restrict__ out, long long n, int C) {
  long long i = (long long)blockIdx.x * blockDim.x + threadIdx.x;
  if (i >= n) return;
  float v = acc[i] + bias[(int)(i % C)];
  out[i] = (_Float16)(v > 0.f ? v : 0.f);
}

// relu(acc + bias) in place (f32)
__global__ void finalize_relu_f32(float* __restrict__ acc, const float* __restrict__ bias,
                                  long long n, int C) {
  long long i = (long long)blockIdx.x * blockDim.x + threadIdx.x;
  if (i >= n) return;
  float v = acc[i] + bias[(int)(i % C)];
  acc[i] = v > 0.f ? v : 0.f;
}

// scatter h2[n] -> xseq16[batch[n]*T + time[n]]
__global__ void scatter_seq(const float* __restrict__ h2, const int* __restrict__ batch,
                            const int* __restrict__ ntime, _Float16* __restrict__ xseq,
                            int Nn, int T) {
  long long idx = (long long)blockIdx.x * blockDim.x + threadIdx.x;
  if (idx >= (long long)Nn * 16) return;
  int n = (int)(idx >> 4);
  int q = (int)(idx & 15) * 4;
  long long row = (long long)batch[n] * T + ntime[n];
  v4h o;
#pragma unroll
  for (int j = 0; j < 4; ++j) o[j] = (_Float16)h2[(long long)n * 64 + q + j];
  *reinterpret_cast<v4h*>(xseq + row * 64 + q) = o;
}

// helper: build a v16h A-fragment from two contiguous 16B half-chunks
__device__ __forceinline__ v16h load_afrag(const _Float16* ap) {
  v8h lo = *reinterpret_cast<const v8h*>(ap);        // K {0..7}+16hi
  v8h hs = *reinterpret_cast<const v8h*>(ap + 8);    // K {8..15}+16hi
  v16h a;
#pragma unroll
  for (int j = 0; j < 8; ++j) { a[j] = lo[j]; a[8 + j] = hs[j]; }
  return a;
}

// ---------------------------------------------------------------------------
// WMMA GEMM, compile-time shape: C[M,Ntot] = A(f16,[M, KC*32]) * B + biases
//   TB==0: B[k][n] = W[k*Ntot+n]     TB==1: B[k][n] = W[n*K+k]
// block = 128 threads (4 waves); each wave: one M-tile (grid-stride) x
// NT*16 columns at blockIdx.y*NT*16. B fragments register-resident; only a
// single A fragment live at a time (kc outer loop) -> no spills.
// ---------------------------------------------------------------------------
template <int KC, int NT, int TB>
__global__ void wmma_gemm(const _Float16* __restrict__ A, const float* __restrict__ W,
                          const float* __restrict__ bias1, const float* __restrict__ bias2,
                          float* __restrict__ C, int M, int Ntot) {
  constexpr int K = KC * 32;
  const int lane = threadIdx.x & 31;
  const int wave = threadIdx.x >> 5;
  const int r = lane & 15, hi = lane >> 4;
  const int colbase = blockIdx.y * (NT * 16);

  // B fragments (contiguous k = 16*hi + s, same order as load_afrag: lane = column, K pairs 2i+16*hi)
  v16h bf[NT][KC];
#pragma unroll
  for (int nt = 0; nt < NT; ++nt) {
    const int col = colbase + nt * 16 + r;
#pragma unroll
    for (int kc = 0; kc < KC; ++kc) {
#pragma unroll
      for (int i = 0; i < 8; ++i) {
        const int k = kc * 32 + 2 * i + 16 * hi;
        float w0, w1;
        if (TB) { w0 = W[(long long)col * K + k];    w1 = W[(long long)col * K + k + 1]; }
        else    { w0 = W[(long long)k * Ntot + col]; w1 = W[(long long)(k + 1) * Ntot + col]; }
        bf[nt][kc][2 * i]     = (_Float16)w0;
        bf[nt][kc][2 * i + 1] = (_Float16)w1;
      }
    }
  }

  const int Mtiles = (M + 15) >> 4;
  for (int mt = blockIdx.x * 4 + wave; mt < Mtiles; mt += gridDim.x * 4) {
    if (mt + (int)gridDim.x * 4 < Mtiles)
      __builtin_prefetch(A + (long long)(mt + gridDim.x * 4) * 16 * K, 0, 1);
    int row = mt * 16 + r; if (row >= M) row = M - 1;
    const _Float16* abase = A + (long long)row * K + 16 * hi;

    v8f acc[NT];
#pragma unroll
    for (int nt = 0; nt < NT; ++nt) acc[nt] = v8f_zero();

#pragma unroll
    for (int kc = 0; kc < KC; ++kc) {
      const v16h a = load_afrag(abase + kc * 32);
#pragma unroll
      for (int nt = 0; nt < NT; ++nt)
        acc[nt] = __builtin_amdgcn_wmma_f32_16x16x32_f16(
            false, a, false, bf[nt][kc], (short)0, acc[nt], false, false);
    }
#pragma unroll
    for (int nt = 0; nt < NT; ++nt) {
      const int col = colbase + nt * 16 + r;
      const float b = (bias1 ? bias1[col] : 0.f) + (bias2 ? bias2[col] : 0.f);
#pragma unroll
      for (int i = 0; i < 8; ++i) {
        const int orow = mt * 16 + i + 8 * hi;          // ISA C/D layout
        if (orow < M) C[(long long)orow * Ntot + col] = acc[nt][i] + b;
      }
    }
  }
}

// ---------------------------------------------------------------------------
// Persistent single-workgroup LSTM recurrence (one layer), T = 50 constant.
//   G   : [256*T, 256] precomputed x@Wih^T + bih + bhh, row = b*T + t
//   Whh : [256, 64]
// h state (32KB) and Whh^T (32KB, f16) staged in LDS; c state in registers.
// 1024 threads = 32 waves; wave (wq, cg): M-tiles {wq, wq+8}, cols
// 16*cg..16*cg+15 within each of the 4 gates (gate math register-local).
// ---------------------------------------------------------------------------
__global__ __launch_bounds__(1024, 1) void lstm_recurrent(
    const float* __restrict__ G, const float* __restrict__ Whh,
    _Float16* __restrict__ y16, float* __restrict__ hn) {
  constexpr int T = 50;
  __shared__ _Float16 hbuf[256 * 64];
  __shared__ _Float16 wbuf[256 * 64];     // Whh as f16, same [col][k] layout
  const int tid = threadIdx.x;
  const int lane = tid & 31, wave = tid >> 5;
  const int r = lane & 15, hi = lane >> 4;
  const int wq = wave >> 2;
  const int n0 = (wave & 3) * 16;

  for (int i = tid; i < 256 * 64; i += 1024) {
    hbuf[i] = (_Float16)0.f;
    wbuf[i] = (_Float16)Whh[i];
  }

  v8f cst[2];
  cst[0] = v8f_zero();
  cst[1] = v8f_zero();
  __syncthreads();

  for (int t = 0; t < T; ++t) {
    v8f hout[2];
#pragma unroll
    for (int ph = 0; ph < 2; ++ph) {
      const int mt = wq + ph * 8;
      // A fragments from LDS h state (each = two contiguous 16B chunks)
      const _Float16* ab = hbuf + (mt * 16 + r) * 64 + 16 * hi;
      const v16h af0 = load_afrag(ab);
      const v16h af1 = load_afrag(ab + 32);
      // G base: rows differ by T*256 floats, gates by 64 floats -> imm offsets
      const float* gb = G + ((long long)(mt * 16 + 8 * hi) * T + t) * 256 + n0 + r;
      v8f ga[4];
#pragma unroll
      for (int g = 0; g < 4; ++g) {
        v8f a;
#pragma unroll
        for (int i = 0; i < 8; ++i) a[i] = gb[i * (T * 256) + g * 64];
        // B fragments from LDS Whh^T: 32 contiguous bytes per lane
        const _Float16* wb = wbuf + (g * 64 + n0 + r) * 64 + 16 * hi;
        const v16h b0 = *reinterpret_cast<const v16h*>(wb);
        const v16h b1 = *reinterpret_cast<const v16h*>(wb + 32);
        a = __builtin_amdgcn_wmma_f32_16x16x32_f16(false, af0, false, b0,
                                                   (short)0, a, false, false);
        a = __builtin_amdgcn_wmma_f32_16x16x32_f16(false, af1, false, b1,
                                                   (short)0, a, false, false);
        ga[g] = a;
      }
      // gate math (PyTorch order i,f,g,o), c in registers
#pragma unroll
      for (int i = 0; i < 8; ++i) {
        float ig = sigmoidf_(ga[0][i]);
        float fg = sigmoidf_(ga[1][i]);
        float gg = tanhf_(ga[2][i]);
        float og = sigmoidf_(ga[3][i]);
        float c = fg * cst[ph][i] + ig * gg;
        cst[ph][i] = c;
        hout[ph][i] = og * tanhf_(c);
      }
    }
    __syncthreads();                       // all reads of h(t-1) complete
#pragma unroll
    for (int ph = 0; ph < 2; ++ph) {
      const int mt = wq + ph * 8;
      const int u = n0 + r;
      _Float16* hb = hbuf + (mt * 16 + 8 * hi) * 64 + u;
      _Float16* yb = y16 ? y16 + ((long long)(mt * 16 + 8 * hi) * T + t) * 64 + u : nullptr;
      float*    nb = hn  ? hn + (mt * 16 + 8 * hi) * 64 + u : nullptr;
#pragma unroll
      for (int i = 0; i < 8; ++i) {
        _Float16 hv = (_Float16)hout[ph][i];
        hb[i * 64] = hv;
        if (y16) yb[(long long)i * (T * 64)] = hv;
        if (hn && t == T - 1) nb[i * 64] = hout[ph][i];
      }
    }
    __syncthreads();                       // h(t) visible before next step
  }
}

// ---------------------------------------------------------------------------
// final FC: out[b,j] = hn[b,:] . fcW[j,:] + fcb[j]
// ---------------------------------------------------------------------------
__global__ void fc_kernel(const float* __restrict__ hn, const float* __restrict__ fcW,
                          const float* __restrict__ fcb, float* __restrict__ out, int NG) {
  int idx = blockIdx.x * blockDim.x + threadIdx.x;
  if (idx >= NG * 36) return;
  int b = idx / 36, j = idx % 36;
  float s = fcb[j];
#pragma unroll 8
  for (int u = 0; u < 64; ++u) s += hn[b * 64 + u] * fcW[j * 64 + u];
  out[idx] = s;
}

// ---------------------------------------------------------------------------
extern "C" void kernel_launch(void* const* d_in, const int* in_sizes, int n_in,
                              void* d_out, int out_size, void* d_ws, size_t ws_size,
                              hipStream_t stream) {
  (void)n_in; (void)out_size; (void)ws_size;
  const float* x      = (const float*)d_in[0];
  const float* W1     = (const float*)d_in[1];
  const float* a_src1 = (const float*)d_in[2];
  const float* a_dst1 = (const float*)d_in[3];
  const float* b1     = (const float*)d_in[4];
  const float* W2     = (const float*)d_in[5];
  const float* a_src2 = (const float*)d_in[6];
  const float* a_dst2 = (const float*)d_in[7];
  const float* b2     = (const float*)d_in[8];
  const float* Wih0   = (const float*)d_in[9];
  const float* Whh0   = (const float*)d_in[10];
  const float* bih0   = (const float*)d_in[11];
  const float* bhh0   = (const float*)d_in[12];
  const float* Wih1   = (const float*)d_in[13];
  const float* Whh1   = (const float*)d_in[14];
  const float* bih1   = (const float*)d_in[15];
  const float* bhh1   = (const float*)d_in[16];
  const float* fcW    = (const float*)d_in[17];
  const float* fcb    = (const float*)d_in[18];
  const int*   ei     = (const int*)d_in[19];
  const int*   batch  = (const int*)d_in[20];
  const int*   ntime  = (const int*)d_in[21];

  const int N  = in_sizes[20];
  const int E  = in_sizes[19] / 2;
  const int EP = E + N;                      // with self loops
  const int T  = 50, NG = 256;

  // --- workspace layout (regions reused across phases) ---
  char* w = (char*)d_ws;
  const size_t R0 = (size_t)N * 128 * 4;     // xw1, later h1f16 overlay
  float*    xw1    = (float*)w;
  _Float16* h1f16  = (_Float16*)w;           // overlays xw1 (xw1 dead by then)
  float*    acc1   = (float*)(w + R0);       // N*128 f32
  float*    xw2    = acc1;                   // reuse (first N*64 f32)
  float*    acc2   = (float*)(w + R0 + (size_t)N * 64 * 4);
  char* sm = w + 2 * R0;
  float*    es1    = (float*)sm;  sm += (size_t)N * 2 * 4;
  float*    ed1    = (float*)sm;  sm += (size_t)N * 2 * 4;
  unsigned* emax1  = (unsigned*)sm; sm += (size_t)N * 2 * 4;
  float*    denom1 = (float*)sm;  sm += (size_t)N * 2 * 4;
  float*    ebuf   = (float*)sm;  sm += (size_t)EP * 2 * 4;
  _Float16* xseq16 = (_Float16*)sm; sm += (size_t)NG * T * 64 * 2;
  float*    G0     = (float*)sm;  sm += (size_t)NG * T * 256 * 4;
  _Float16* y16    = (_Float16*)sm; sm += (size_t)NG * T * 64 * 2;
  float*    G1     = (float*)sm;  sm += (size_t)NG * T * 256 * 4;
  float*    hn     = (float*)sm;
  // layer-2 aliases (sizes are strictly smaller)
  float* es2 = es1; float* ed2 = ed1; unsigned* emax2 = emax1;
  float* denom2 = denom1; float* ebuf2 = ebuf;

  auto zero = [&](void* p, long long words) {
    zero_u32<<<dim3(2048), dim3(256), 0, stream>>>((unsigned*)p, words);
  };

  // ===== GAT layer 1 =====
  zero(acc1,  (long long)N * 128);
  zero(emax1, (long long)N * 2);
  zero(denom1,(long long)N * 2);
  gat1_node<<<dim3((N + 7) / 8), dim3(256), 0, stream>>>(x, W1, a_src1, a_dst1,
                                                         xw1, es1, ed1, N);
  {
    long long it = (long long)EP * 2;
    edge_pass1<2><<<dim3((unsigned)((it + 255) / 256)), dim3(256), 0, stream>>>(
        ei, E, EP, es1, ed1, ebuf, emax1);
    edge_pass2<2><<<dim3((unsigned)((it + 255) / 256)), dim3(256), 0, stream>>>(
        ei, E, EP, ebuf, emax1, denom1);
    long long it3 = (long long)EP * 32;
    edge_pass3<2, 128, 32><<<dim3((unsigned)((it3 + 255) / 256)), dim3(256), 0, stream>>>(
        ei, E, EP, ebuf, denom1, xw1, acc1);
  }
  finalize_relu_f16<<<dim3((unsigned)(((long long)N * 128 + 255) / 256)), dim3(256), 0, stream>>>(
      acc1, b1, h1f16, (long long)N * 128, 128);

  // ===== GAT layer 2: xw2 = h1 @ W2 (WMMA, K=128, 2 col-tiles/wave) =====
  {
    int mtiles = (N + 15) / 16;
    wmma_gemm<4, 2, 0><<<dim3((mtiles + 3) / 4, 2), dim3(128), 0, stream>>>(
        h1f16, W2, nullptr, nullptr, xw2, N, 64);
  }
  gat2_scores<<<dim3((N + 7) / 8), dim3(256), 0, stream>>>(xw2, a_src2, a_dst2, es2, ed2, N);
  zero(emax2,  (long long)N);
  zero(denom2, (long long)N);
  zero(acc2,   (long long)N * 64);
  {
    long long it = (long long)EP;
    edge_pass1<1><<<dim3((unsigned)((it + 255) / 256)), dim3(256), 0, stream>>>(
        ei, E, EP, es2, ed2, ebuf2, emax2);
    edge_pass2<1><<<dim3((unsigned)((it + 255) / 256)), dim3(256), 0, stream>>>(
        ei, E, EP, ebuf2, emax2, denom2);
    long long it3 = (long long)EP * 16;
    edge_pass3<1, 64, 16><<<dim3((unsigned)((it3 + 255) / 256)), dim3(256), 0, stream>>>(
        ei, E, EP, ebuf2, denom2, xw2, acc2);
  }
  finalize_relu_f32<<<dim3((unsigned)(((long long)N * 64 + 255) / 256)), dim3(256), 0, stream>>>(
      acc2, b2, (long long)N * 64, 64);

  // ===== temporal scatter =====
  zero(xseq16, (long long)NG * T * 32);   // halves/2 = u32 words
  scatter_seq<<<dim3((unsigned)(((long long)N * 16 + 255) / 256)), dim3(256), 0, stream>>>(
      acc2, batch, ntime, xseq16, N, T);

  // ===== LSTM layer 0 =====
  wmma_gemm<2, 2, 1><<<dim3(200, 8), dim3(128), 0, stream>>>(
      xseq16, Wih0, bih0, bhh0, G0, NG * T, 256);
  lstm_recurrent<<<dim3(1), dim3(1024), 0, stream>>>(G0, Whh0, y16, nullptr);

  // ===== LSTM layer 1 =====
  wmma_gemm<2, 2, 1><<<dim3(200, 8), dim3(128), 0, stream>>>(
      y16, Wih1, bih1, bhh1, G1, NG * T, 256);
  lstm_recurrent<<<dim3(1), dim3(1024), 0, stream>>>(G1, Whh1, nullptr, hn);

  // ===== FC head =====
  fc_kernel<<<dim3((NG * 36 + 255) / 256), dim3(256), 0, stream>>>(
      hn, fcW, fcb, (float*)d_out, NG);
}
